// SingleHeadAttention_61220463837737
// MI455X (gfx1250) — matrix operand order repaired and checked by reading, not compiled, not run on hardware
//
#include <hip/hip_runtime.h>

// ---------------------------------------------------------------------------
// MI455X (gfx1250) single-head causal attention.
// Compute-bound (~30 GFLOP matmul vs 134 MB @ 23.3 TB/s) -> all matmuls via
// v_wmma_f32_16x16x32_bf16 (f32 accumulate), wave32.
// Attention: K/V key blocks staged in LDS once per 8-wave block through
// GLOBAL_LOAD_ASYNC_TO_LDS_B128 (ASYNCcnt), double-buffered so the DMA of
// block kb+1 overlaps the WMMA+softmax of block kb.
// ---------------------------------------------------------------------------

#define NB   8
#define SEQ  4096
#define HID  1024
#define HD   64

typedef __attribute__((ext_vector_type(16))) __bf16       v16bf;
typedef __attribute__((ext_vector_type(8)))  float        v8f;
typedef __attribute__((ext_vector_type(4)))  unsigned int u32x4;

union ABFrag { u32x4 u[2]; v16bf v; };

__device__ __forceinline__ v8f wmma_bf16(v16bf a, v16bf b, v8f c) {
    // 8 args: (neg_a, A, neg_b, B, c_mod, C, reuse_a, reuse_b)
    return __builtin_amdgcn_wmma_f32_16x16x32_bf16(false, a, false, b,
                                                   (short)0, c, false, false);
}

// A-fragment (16x32 bf16, row-major source, row stride ld elements).
// ISA 7.12.2: lane L holds row M=L&15; lanes 0-15 K=0..7 & 16..23,
// lanes 16-31 K=8..15 & 24..31 -> two contiguous 16B chunks per lane.
__device__ __forceinline__ v16bf load_a_bf16(const __bf16* src, int ld, int lane) {
    const int half = lane >> 4, m = lane & 15;
    const __bf16* p = src + (long)m * ld + half * 8;
    ABFrag f;
    f.u[0] = *(const u32x4*)(p);
    f.u[1] = *(const u32x4*)(p + 16);
    return f.v;
}

// A-fragment with in-register f32 -> bf16 conversion (for x which is f32).
__device__ __forceinline__ v16bf load_a_f32(const float* src, int ld, int lane) {
    const int half = lane >> 4, m = lane & 15;
    const float* p = src + (long)m * ld + half * 8;
    float4 a0 = *(const float4*)(p);
    float4 a1 = *(const float4*)(p + 4);
    float4 a2 = *(const float4*)(p + 16);
    float4 a3 = *(const float4*)(p + 20);
    v16bf r;
    r[0]  = (__bf16)a0.x; r[1]  = (__bf16)a0.y; r[2]  = (__bf16)a0.z; r[3]  = (__bf16)a0.w;
    r[4]  = (__bf16)a1.x; r[5]  = (__bf16)a1.y; r[6]  = (__bf16)a1.z; r[7]  = (__bf16)a1.w;
    r[8]  = (__bf16)a2.x; r[9]  = (__bf16)a2.y; r[10] = (__bf16)a2.z; r[11] = (__bf16)a2.w;
    r[12] = (__bf16)a3.x; r[13] = (__bf16)a3.y; r[14] = (__bf16)a3.z; r[15] = (__bf16)a3.w;
    return r;
}

// B-fragment (32x16 bf16) from memory where the K dimension is contiguous per
// column: element (k, n) at src[(long)n*ldn + k].
// ISA 7.12.2: lane L holds column N=L&15; lanes 0-15 K=0..7 & 16..23,
// lanes 16-31 K=8..15 & 24..31 -> two contiguous 16B chunks per lane.
__device__ __forceinline__ v16bf load_b_kcontig(const __bf16* src, long ldn, int lane) {
    const int half = lane >> 4, n = lane & 15;
    const __bf16* p = src + (long)n * ldn + half * 8;
    ABFrag f;
    f.u[0] = *(const u32x4*)(p);
    f.u[1] = *(const u32x4*)(p + 16);
    return f.v;
}

__device__ __forceinline__ float half_red_max(float v) {
    v = fmaxf(v, __shfl_xor(v, 1));
    v = fmaxf(v, __shfl_xor(v, 2));
    v = fmaxf(v, __shfl_xor(v, 4));
    v = fmaxf(v, __shfl_xor(v, 8));   // masks <16: stay within 16-lane half
    return v;
}
__device__ __forceinline__ float half_red_sum(float v) {
    v += __shfl_xor(v, 1);
    v += __shfl_xor(v, 2);
    v += __shfl_xor(v, 4);
    v += __shfl_xor(v, 8);
    return v;
}

// ---- CDNA5 async global->LDS copy (ASYNCcnt-tracked, bypasses VGPRs) ------
// Generic pointers to LDS carry the byte offset in their low 32 bits
// (aperture addressing).
__device__ __forceinline__ void async_load_b128(unsigned lds_byte_off, const void* gaddr) {
    asm volatile("global_load_async_to_lds_b128 %0, %1, off"
                 :: "v"(lds_byte_off), "v"(gaddr)
                 : "memory");
}
__device__ __forceinline__ void wait_async_0() {
#if __has_builtin(__builtin_amdgcn_s_wait_asynccnt)
    __builtin_amdgcn_s_wait_asynccnt(0);
#else
    asm volatile("s_wait_asynccnt 0x0" ::: "memory");
#endif
}
__device__ __forceinline__ void wait_async_4() {
#if __has_builtin(__builtin_amdgcn_s_wait_asynccnt)
    __builtin_amdgcn_s_wait_asynccnt(4);
#else
    asm volatile("s_wait_asynccnt 0x4" ::: "memory");
#endif
}
__device__ __forceinline__ unsigned lds_off(const void* p) {
    return (unsigned)(unsigned long long)p;
}

// ---------------------------------------------------------------------------
// Kernel 0: W [H][D] f32 -> Wt [3][D][H] bf16 (transpose so K(=h) contiguous)
// ---------------------------------------------------------------------------
__global__ void convert_w_kernel(const float* __restrict__ Wq,
                                 const float* __restrict__ Wk,
                                 const float* __restrict__ Wv,
                                 __bf16* __restrict__ Wt) {
    int idx = blockIdx.x * blockDim.x + threadIdx.x;     // [0, 3*HD*HID)
    int p   = idx / (HD * HID);
    int rem = idx % (HD * HID);
    int d = rem / HID, h = rem % HID;
    const float* W = (p == 0) ? Wq : ((p == 1) ? Wk : Wv);
    Wt[idx] = (__bf16)W[(long)h * HD + d];
}

// ---------------------------------------------------------------------------
// Kernel 1: QKV projection. Per wave: 16 rows x D=64, K=1024.
// Q,K -> row-major [B][S][HD] bf16 ; V -> transposed [B][HD][S] bf16.
// ---------------------------------------------------------------------------
__global__ __launch_bounds__(256)
void qkv_proj_kernel(const float* __restrict__ x,
                     const __bf16* __restrict__ Wt,
                     const float* __restrict__ bq,
                     const float* __restrict__ bk,
                     const float* __restrict__ bv,
                     __bf16* __restrict__ Qb,
                     __bf16* __restrict__ Kb,
                     __bf16* __restrict__ Vt) {
    const int wave = (blockIdx.x * blockDim.x + threadIdx.x) >> 5; // 0..2047
    const int lane = threadIdx.x & 31;
    const int b    = wave / (SEQ / 16);
    const int row0 = (wave % (SEQ / 16)) * 16;
    const float* xrow = x + ((long)b * SEQ + row0) * HID;
    const int half = lane >> 4, n = lane & 15;

#pragma unroll
    for (int p = 0; p < 3; ++p) {
        const __bf16* W    = Wt + (long)p * HD * HID;
        const float*  bias = (p == 0) ? bq : ((p == 1) ? bk : bv);
        v8f acc[4] = {{0,0,0,0,0,0,0,0},{0,0,0,0,0,0,0,0},
                      {0,0,0,0,0,0,0,0},{0,0,0,0,0,0,0,0}};
        for (int kk = 0; kk < HID; kk += 32) {
            v16bf a = load_a_f32(xrow + kk, HID, lane);
#pragma unroll
            for (int t = 0; t < 4; ++t) {
                v16bf bf = load_b_kcontig(W + (long)(t * 16) * HID + kk, HID, lane);
                acc[t] = wmma_bf16(a, bf, acc[t]);
            }
        }
        // Bias add + store.  C layout: lane holds (M = r + 8*half, N = n).
#pragma unroll
        for (int t = 0; t < 4; ++t) {
            float bval = bias[t * 16 + n];
#pragma unroll
            for (int r = 0; r < 8; ++r) {
                float v = acc[t][r] + bval;
                int   M = row0 + r + 8 * half;
                int   d = t * 16 + n;
                if (p == 0)      Qb[((long)b * SEQ + M) * HD + d] = (__bf16)v;
                else if (p == 1) Kb[((long)b * SEQ + M) * HD + d] = (__bf16)v;
                else             Vt[((long)b * HD + d) * SEQ + M] = (__bf16)v;
            }
        }
    }
}

// ---------------------------------------------------------------------------
// Kernel 2: causal flash attention.
// Block = 8 waves = 8 consecutive 16-row query tiles of one batch.
// K/V key blocks (64 keys) staged in LDS via async DMA, double-buffered,
// shared by all 8 waves.  Per wave: 16 WMMAs + online softmax per key block.
// ---------------------------------------------------------------------------
__global__ __launch_bounds__(256)
void attn_kernel(const __bf16* __restrict__ Qb,
                 const __bf16* __restrict__ Kb,
                 const __bf16* __restrict__ Vt,
                 float* __restrict__ out) {
    __shared__ __align__(16) unsigned short sK[2][64 * 64];  // [key][d], 8KB each
    __shared__ __align__(16) unsigned short sV[2][64 * 64];  // [d][key], 8KB each
    __shared__ __align__(16) unsigned short sP[8][16 * 64];  // per-wave P tiles

    const int tid  = threadIdx.x;
    const int wib  = tid >> 5;
    const int lane = tid & 31;
    const int b      = blockIdx.x / (SEQ / 128);
    const int q0_blk = (blockIdx.x % (SEQ / 128)) * 128;
    const int q0     = q0_blk + wib * 16;
    const int half   = lane >> 4, n = lane & 15;

    const __bf16* Q = Qb + ((long)b * SEQ + q0) * HD;
    const __bf16* K = Kb + (long)b * SEQ * HD;
    const __bf16* V = Vt + (long)b * HD * SEQ;
    __bf16* myP = (__bf16*)&sP[wib][0];

    // Q A-fragments for d = 0..31 and 32..63 (reused across all key blocks).
    v16bf qa0 = load_a_bf16(Q, HD, lane);
    v16bf qa1 = load_a_bf16(Q + 32, HD, lane);

    v8f o[4] = {{0,0,0,0,0,0,0,0},{0,0,0,0,0,0,0,0},
                {0,0,0,0,0,0,0,0},{0,0,0,0,0,0,0,0}};
    float mrow[8], lrow[8];
#pragma unroll
    for (int r = 0; r < 8; ++r) { mrow[r] = -__builtin_inff(); lrow[r] = 0.f; }

    // Issue one K tile (contiguous 8KB) + one V tile (64 rows x 128B) async.
    // 2 + 2 chunks of 16B per thread -> 4 ASYNC instructions per wave.
    auto issue_tile = [&](int kb, int buf) {
        const int k0 = kb * 64;
        const char* kp = (const char*)(K + (long)k0 * HD);
        const unsigned ldsK = lds_off(&sK[buf][0]);
        async_load_b128(ldsK + tid * 16,         kp + tid * 16);
        async_load_b128(ldsK + (tid + 256) * 16, kp + (tid + 256) * 16);
        const char* vp = (const char*)(V + k0);
        const unsigned ldsV = lds_off(&sV[buf][0]);
        const int c0 = tid, c1 = tid + 256;
        async_load_b128(ldsV + c0 * 16, vp + (long)(c0 >> 3) * SEQ * 2 + (c0 & 7) * 16);
        async_load_b128(ldsV + c1 * 16, vp + (long)(c1 >> 3) * SEQ * 2 + (c1 & 7) * 16);
    };

    const int nblocks_w   = (q0 + 16 + 63) / 64;           // this wave's range
    const int nblocks_blk = (q0_blk + 128 + 63) / 64;      // block-wide range

    issue_tile(0, 0);
    for (int kb = 0; kb < nblocks_blk; ++kb) {
        const int buf = kb & 1;
        const int k0  = kb * 64;
        if (kb + 1 < nblocks_blk) {
            issue_tile(kb + 1, buf ^ 1);   // prev readers of buf^1 done (barrier kb-1)
            wait_async_4();                // only the 4 new ones may be outstanding
        } else {
            wait_async_0();
        }
        __syncthreads();                   // tile kb visible to all waves

        if (kb < nblocks_w) {              // wave-uniform -> EXEC stays all-1s
            const __bf16* sKp = (const __bf16*)&sK[buf][0];
            const __bf16* sVp = (const __bf16*)&sV[buf][0];
            // ---- S = Q K^T (16x64) : 8 WMMAs, B frags from LDS ----
            v8f s[4] = {{0,0,0,0,0,0,0,0},{0,0,0,0,0,0,0,0},
                        {0,0,0,0,0,0,0,0},{0,0,0,0,0,0,0,0}};
#pragma unroll
            for (int t = 0; t < 4; ++t) {
                v16bf b0 = load_b_kcontig(sKp + (t * 16) * 64,      64, lane);
                v16bf b1 = load_b_kcontig(sKp + (t * 16) * 64 + 32, 64, lane);
                s[t] = wmma_bf16(qa0, b0, s[t]);
                s[t] = wmma_bf16(qa1, b1, s[t]);
            }
            // ---- causal mask (pre-scale, -inf) + scale 1/sqrt(64) ----
#pragma unroll
            for (int t = 0; t < 4; ++t)
#pragma unroll
                for (int r = 0; r < 8; ++r) {
                    int key = k0 + t * 16 + n;
                    int qr  = q0 + r + 8 * half;
                    float sv = s[t][r] * 0.125f;
                    s[t][r] = (key <= qr) ? sv : -__builtin_inff();
                }
            // ---- online softmax (a row spans 16 lanes of one half-wave) ----
            float mnew[8], alpha[8];
#pragma unroll
            for (int r = 0; r < 8; ++r) {
                float bm = fmaxf(fmaxf(s[0][r], s[1][r]), fmaxf(s[2][r], s[3][r]));
                bm = half_red_max(bm);
                mnew[r]  = fmaxf(mrow[r], bm);
                alpha[r] = __expf(mrow[r] - mnew[r]);
                mrow[r]  = mnew[r];
            }
#pragma unroll
            for (int r = 0; r < 8; ++r) {
                float rs = 0.f;
#pragma unroll
                for (int t = 0; t < 4; ++t) {
                    float pv = __expf(s[t][r] - mnew[r]);
                    s[t][r] = pv;
                    rs += pv;
                }
                rs = half_red_sum(rs);
                lrow[r] = lrow[r] * alpha[r] + rs;
#pragma unroll
                for (int t = 0; t < 4; ++t) o[t][r] *= alpha[r];
            }
            // ---- P: C-layout -> A-layout via this wave's LDS tile ----
#pragma unroll
            for (int t = 0; t < 4; ++t)
#pragma unroll
                for (int r = 0; r < 8; ++r)
                    myP[(r + 8 * half) * 64 + t * 16 + n] = (__bf16)s[t][r];
            v16bf pa0 = load_a_bf16(myP,      64, lane);   // keys k0+0..31
            v16bf pa1 = load_a_bf16(myP + 32, 64, lane);   // keys k0+32..63
            // ---- O += P V : 8 WMMAs, B frags from LDS V tile ----
#pragma unroll
            for (int t = 0; t < 4; ++t) {
                v16bf vb0 = load_b_kcontig(sVp + (t * 16) * 64,      64, lane);
                v16bf vb1 = load_b_kcontig(sVp + (t * 16) * 64 + 32, 64, lane);
                o[t] = wmma_bf16(pa0, vb0, o[t]);
                o[t] = wmma_bf16(pa1, vb1, o[t]);
            }
        }
        __syncthreads();                   // everyone done reading buf
    }
    // ---- epilogue: O / l ----
#pragma unroll
    for (int t = 0; t < 4; ++t)
#pragma unroll
        for (int r = 0; r < 8; ++r) {
            int M = q0 + r + 8 * half;
            out[((long)b * SEQ + M) * HD + t * 16 + n] = o[t][r] / lrow[r];
        }
}

// ---------------------------------------------------------------------------
extern "C" void kernel_launch(void* const* d_in, const int* in_sizes, int n_in,
                              void* d_out, int out_size, void* d_ws, size_t ws_size,
                              hipStream_t stream) {
    const float* x  = (const float*)d_in[0];
    const float* Wq = (const float*)d_in[1];
    const float* bq = (const float*)d_in[2];
    const float* Wk = (const float*)d_in[3];
    const float* bk = (const float*)d_in[4];
    const float* Wv = (const float*)d_in[5];
    const float* bv = (const float*)d_in[6];
    float* out = (float*)d_out;

    // Workspace layout (bf16): Wt[3][HD][HID] | Q[B][S][HD] | K[B][S][HD] | Vt[B][HD][S]
    __bf16* wsb = (__bf16*)d_ws;
    __bf16* Wt = wsb;
    __bf16* Qb = Wt + (long)3 * HD * HID;
    __bf16* Kb = Qb + (long)NB * SEQ * HD;
    __bf16* Vt = Kb + (long)NB * SEQ * HD;

    convert_w_kernel<<<(3 * HD * HID) / 256, 256, 0, stream>>>(Wq, Wk, Wv, Wt);

    // B*S/16 = 2048 wave-tiles, 8 waves per 256-thread block -> 256 blocks
    qkv_proj_kernel<<<256, 256, 0, stream>>>(x, Wt, bq, bk, bv, Qb, Kb, Vt);
    attn_kernel<<<256, 256, 0, stream>>>(Qb, Kb, Vt, out);
}
